// CannyEdgeLoss_34780645163474
// MI455X (gfx1250) — compile-verified, matrix operand order checked by reading, not run on hardware
//
#include <hip/hip_runtime.h>
#include <hip/hip_bf16.h>
#include <math.h>

#ifndef __has_builtin
#define __has_builtin(x) 0
#endif

typedef __attribute__((ext_vector_type(16))) _Float16 v16h;
typedef __attribute__((ext_vector_type(8)))  float    v8f;

/* exact pointer types for the gfx1250 async global->LDS builtins */
typedef __attribute__((address_space(1))) int   g_int;
typedef __attribute__((address_space(3))) int   l_int;
typedef __attribute__((address_space(1))) char  g_char;
typedef __attribute__((address_space(3))) char  l_char;

#define IMG_B 32
#define IMG_H 512
#define IMG_W 512
#define HW    (IMG_H*IMG_W)            /* 262144  */
#define BHW   (IMG_B*HW)               /* 8388608 */
#define WPR   (IMG_W/32)               /* 16 mask words per row */
#define WPI   (IMG_B*IMG_H*WPR)        /* 262144 words per image */

/* ---- workspace layout (bytes); total ~25.2 MB ---- */
#define ACC_OFF     0
#define FLAG_OFF    8
#define G8_OFF      64
#define WEAK_OFF    (G8_OFF + 2*BHW)
#define STRONG_OFF  (WEAK_OFF + 2*WPI*4)
#define CUR_OFF     (STRONG_OFF + 2*WPI*4)
#define NXT_OFF     (CUR_OFF + 2*WPI*4)

#define TG22F 0.4142135623730951f
#define TG67F 2.414213562373095f

/* ================= CDNA5 async global->LDS helpers ================= */
__device__ __forceinline__ void async_g2l_b32(unsigned int* lds_dst, const unsigned int* gsrc) {
#if __has_builtin(__builtin_amdgcn_global_load_async_to_lds_b32)
  __builtin_amdgcn_global_load_async_to_lds_b32(
      (g_int*)gsrc, (l_int*)lds_dst, 0, 0);
#else
  *lds_dst = *gsrc;   /* fallback: plain copy through VGPRs */
#endif
}

__device__ __forceinline__ void async_g2l_b8(unsigned char* lds_dst, const unsigned char* gsrc) {
#if __has_builtin(__builtin_amdgcn_global_load_async_to_lds_b8)
  __builtin_amdgcn_global_load_async_to_lds_b8(
      (g_char*)gsrc, (l_char*)lds_dst, 0, 0);
#else
  *lds_dst = *gsrc;
#endif
}

__device__ __forceinline__ void wait_async_lds() {
#if __has_builtin(__builtin_amdgcn_s_wait_asynccnt)
  __builtin_amdgcn_s_wait_asynccnt(0);
#elif defined(__AMDGCN__)
  asm volatile("s_wait_asynccnt 0" ::: "memory");
#endif
}

__device__ __forceinline__ int refl(int i, int n) {        /* jnp.pad 'reflect' */
  i = (i < 0) ? -i : i;
  i = (i >= n) ? (2*n - 2 - i) : i;
  return i;
}
__device__ __forceinline__ int iabs(int v) { return v < 0 ? -v : v; }

/* ====================== kernels ====================== */

__global__ __launch_bounds__(64) void k_init(float* acc, unsigned* flags) {
  if (threadIdx.x == 0) { *acc = 0.0f; flags[0] = 0u; flags[1] = 0u; }
}

/* image.max() > 1  <=>  any(v > 1)  */
__global__ __launch_bounds__(256) void k_flag_gt1(const float* __restrict__ x, int n,
                                                  unsigned* __restrict__ flag) {
  int i = blockIdx.x * blockDim.x + threadIdx.x;
  int s = gridDim.x * blockDim.x;
  bool any = false;
  for (; i < n; i += s) any |= (x[i] > 1.0f);
  if (__any((int)any)) {
    if ((threadIdx.x & 31) == 0) atomicOr(flag, 1u);
  }
}

/* luminance -> quantized g, stored as exact uint8 (values 0..255) */
__global__ __launch_bounds__(256) void k_gray(const float* __restrict__ img,
                                              const unsigned* __restrict__ flag,
                                              unsigned char* __restrict__ g8) {
  const float scale = (*flag) ? 0.5f : 1.0f;
  int p = blockIdx.x * blockDim.x + threadIdx.x;
  if (p >= BHW) return;
  int b = p >> 18;             /* HW = 2^18 */
  int q = p & (HW - 1);
  const float* base = img + (size_t)b * 3 * HW + q;
  float r  = base[0]      * scale;
  float gg = base[HW]     * scale;
  float bb = base[2 * HW] * scale;
  float gray = 0.299f * r + 0.587f * gg + 0.114f * bb;
  float gv = floorf(gray * 255.0f);
  gv = fminf(fmaxf(gv, 0.0f), 255.0f);
  g8[p] = (unsigned char)gv;
}

/* Fused Sobel + NMS + double-threshold. block (32,8): one wave per output row
   segment; g tile (36x12, reflect padded) staged via async b8 copies; int16
   mag tile (34x10, zero padded) in LDS; weak/strong bit-packed via ballot. */
__global__ __launch_bounds__(256) void k_sobel_nms(const unsigned char* __restrict__ g8,
                                                   unsigned* __restrict__ weakM,
                                                   unsigned* __restrict__ strongM,
                                                   unsigned* __restrict__ curM) {
  __shared__ unsigned char gsm[12][40];
  __shared__ short ms[10][36];
  const int x0 = blockIdx.x * 32;
  const int y0 = blockIdx.y * 8;
  const int z  = blockIdx.z;                       /* img*32 + batch : 0..63 */
  const unsigned char* gimg = g8 + (size_t)z * HW;
  const int tid = threadIdx.y * 32 + threadIdx.x;

  for (int i = tid; i < 12 * 36; i += 256) {
    int ly = i / 36, lx = i % 36;
    int gy = refl(y0 + ly - 2, IMG_H);
    int gx = refl(x0 + lx - 2, IMG_W);
    async_g2l_b8(&gsm[ly][lx], &gimg[gy * IMG_W + gx]);
  }
  wait_async_lds();
  __syncthreads();

  for (int i = tid; i < 10 * 34; i += 256) {
    int my = i / 34, mx = i % 34;
    int gy = y0 + my - 1, gx = x0 + mx - 1;
    short m = 0;
    if (gy >= 0 && gy < IMG_H && gx >= 0 && gx < IMG_W) {  /* zero-pad for NMS */
      int ly = my + 1, lx = mx + 1;
      int p00 = gsm[ly-1][lx-1], p01 = gsm[ly-1][lx], p02 = gsm[ly-1][lx+1];
      int p10 = gsm[ly  ][lx-1],                      p12 = gsm[ly  ][lx+1];
      int p20 = gsm[ly+1][lx-1], p21 = gsm[ly+1][lx], p22 = gsm[ly+1][lx+1];
      int sgx = (p02 + 2*p12 + p22) - (p00 + 2*p10 + p20);
      int sgy = (p20 + 2*p21 + p22) - (p00 + 2*p01 + p02);
      m = (short)(iabs(sgx) + iabs(sgy));
    }
    ms[my][mx] = m;
  }
  __syncthreads();

  const int tx = threadIdx.x, ty = threadIdx.y;
  int p00 = gsm[ty+1][tx+1], p01 = gsm[ty+1][tx+2], p02 = gsm[ty+1][tx+3];
  int p10 = gsm[ty+2][tx+1],                        p12 = gsm[ty+2][tx+3];
  int p20 = gsm[ty+3][tx+1], p21 = gsm[ty+3][tx+2], p22 = gsm[ty+3][tx+3];
  int sgx = (p02 + 2*p12 + p22) - (p00 + 2*p10 + p20);
  int sgy = (p20 + 2*p21 + p22) - (p00 + 2*p01 + p02);
  int mag = iabs(sgx) + iabs(sgy);
  float ax = (float)iabs(sgx), ay = (float)iabs(sgy);
  bool horiz = ay <= TG22F * ax;
  bool vert  = ay >= TG67F * ax;
  bool d1    = !horiz && !vert && (sgx * sgy >= 0);
  int m1, m2;
  if (horiz)     { m1 = ms[ty+1][tx  ]; m2 = ms[ty+1][tx+2]; }
  else if (vert) { m1 = ms[ty  ][tx+1]; m2 = ms[ty+2][tx+1]; }
  else if (d1)   { m1 = ms[ty  ][tx  ]; m2 = ms[ty+2][tx+2]; }
  else           { m1 = ms[ty  ][tx+2]; m2 = ms[ty+2][tx  ]; }
  bool keep   = (mag > m1) && (mag >= m2);
  bool weak   = keep && (mag > 25);     /* nms > LOW_T  (integer-exact) */
  bool strong = keep && (mag > 76);     /* nms > HIGH_T */

  unsigned wb = (unsigned)__ballot(weak);     /* wave32: 1 bit per pixel */
  unsigned sb = (unsigned)__ballot(strong);
  if (tx == 0) {
    int word = (z * IMG_H + (y0 + ty)) * WPR + blockIdx.x;
    weakM[word] = wb; strongM[word] = sb; curM[word] = sb;
  }
}

/* one hysteresis iteration: nxt = (dilate3x3(cur) & weak) | strong.
   cur rows (16 + 2 halo) staged to LDS via async b32 copies. Masks are
   L2-resident (8 MB total), so each of the 64 iterations is L2-bandwidth. */
__device__ __forceinline__ unsigned hdil(const unsigned* r, int wx) {
  unsigned w = r[wx];
  unsigned l = (wx > 0)  ? r[wx-1] : 0u;
  unsigned q = (wx < 15) ? r[wx+1] : 0u;
  return w | (w << 1) | (l >> 31) | (w >> 1) | (q << 31);
}

__global__ __launch_bounds__(256) void k_hyst(const unsigned* __restrict__ src,
                                              unsigned* __restrict__ dst,
                                              const unsigned* __restrict__ weakM,
                                              const unsigned* __restrict__ strongM) {
  __shared__ unsigned cs[18][16];
  const int y0 = blockIdx.x * 16;
  const int z  = blockIdx.y;                      /* 0..63 */
  const unsigned* s = src + (size_t)z * IMG_H * WPR;
  const int tid = threadIdx.x;

  for (int i = tid; i < 18 * 16; i += 256) {
    int ly = i / 16, wx = i % 16;
    int gy = y0 + ly - 1;
    if (gy >= 0 && gy < IMG_H) async_g2l_b32(&cs[ly][wx], &s[gy * WPR + wx]);
    else                       cs[ly][wx] = 0u;   /* image border: zero pad */
  }
  wait_async_lds();
  __syncthreads();

  const int ty = tid >> 4, wx = tid & 15;
  unsigned dil = hdil(cs[ty], wx) | hdil(cs[ty+1], wx) | hdil(cs[ty+2], wx);
  int gidx = (z * IMG_H + (y0 + ty)) * WPR + wx;
  dst[gidx] = (dil & weakM[gidx]) | strongM[gidx];
}

/* loss = sqrt(popcount(eA ^ eB)). Popcounts accumulate into 16 exact f16
   slots per lane; one v_wmma_f32_16x16x32_f16 against an all-ones B sums all
   512 per-wave partials (every D column equals the 16 row sums). */
__global__ __launch_bounds__(256) void k_loss(const unsigned* __restrict__ eA,
                                              const unsigned* __restrict__ eB,
                                              float* __restrict__ acc) {
  v16h a;
#pragma unroll
  for (int i = 0; i < 16; ++i) a[i] = (_Float16)0.0f;

  const int tid = blockIdx.x * blockDim.x + threadIdx.x;
  const int stride = gridDim.x * blockDim.x;
  int slot = 0;
  for (int i = tid; i < WPI; i += stride) {
    unsigned d = eA[i] ^ eB[i];
    a[slot & 15] += (_Float16)(float)__popc(d);   /* each partial <= 2048: exact */
    ++slot;
  }

  v16h ones;
#pragma unroll
  for (int i = 0; i < 16; ++i) ones[i] = (_Float16)1.0f;
  v8f c = {};
  c = __builtin_amdgcn_wmma_f32_16x16x32_f16(
      /*neg_a=*/false, a, /*neg_b=*/false, ones,
      /*c_mod=*/(short)0, c, /*reuse_a=*/false, /*reuse_b=*/false);

  float s = c[0]+c[1]+c[2]+c[3]+c[4]+c[5]+c[6]+c[7];   /* half the rows */
  s += __shfl_xor(s, 16, 32);                           /* + partner half */
  if ((threadIdx.x & 31) == 0) atomicAdd(acc, s);       /* one add per wave */
}

__global__ __launch_bounds__(64) void k_finish(const float* acc, float* out) {
  if (threadIdx.x == 0) out[0] = sqrtf(*acc);
}

/* ====================== launch ====================== */
extern "C" void kernel_launch(void* const* d_in, const int* in_sizes, int n_in,
                              void* d_out, int out_size, void* d_ws, size_t ws_size,
                              hipStream_t stream) {
  (void)in_sizes; (void)n_in; (void)out_size; (void)ws_size;
  const float* imgA = (const float*)d_in[0];
  const float* imgB = (const float*)d_in[1];
  char* ws = (char*)d_ws;
  float*         acc     = (float*)(ws + ACC_OFF);
  unsigned*      flags   = (unsigned*)(ws + FLAG_OFF);
  unsigned char* g8      = (unsigned char*)(ws + G8_OFF);
  unsigned*      weakM   = (unsigned*)(ws + WEAK_OFF);
  unsigned*      strongM = (unsigned*)(ws + STRONG_OFF);
  unsigned*      curM    = (unsigned*)(ws + CUR_OFF);
  unsigned*      nxtM    = (unsigned*)(ws + NXT_OFF);

  k_init<<<1, 64, 0, stream>>>(acc, flags);

  const int nElems = 3 * BHW;                       /* 25,165,824 */
  k_flag_gt1<<<1024, 256, 0, stream>>>(imgA, nElems, flags + 0);
  k_flag_gt1<<<1024, 256, 0, stream>>>(imgB, nElems, flags + 1);

  k_gray<<<BHW / 256, 256, 0, stream>>>(imgA, flags + 0, g8);
  k_gray<<<BHW / 256, 256, 0, stream>>>(imgB, flags + 1, g8 + BHW);

  k_sobel_nms<<<dim3(IMG_W / 32, IMG_H / 8, 2 * IMG_B), dim3(32, 8), 0, stream>>>(
      g8, weakM, strongM, curM);

  /* fixed 64 iterations == reference while_loop (fixed point after converge) */
  unsigned* pa = curM;
  unsigned* pb = nxtM;
  for (int it = 0; it < 64; ++it) {
    k_hyst<<<dim3(IMG_H / 16, 2 * IMG_B), 256, 0, stream>>>(pa, pb, weakM, strongM);
    unsigned* t = pa; pa = pb; pb = t;              /* pa = freshest */
  }

  k_loss<<<512, 256, 0, stream>>>(pa, pa + WPI, acc);
  k_finish<<<1, 64, 0, stream>>>(acc, (float*)d_out);
}
